// UPRTField_59682865545666
// MI455X (gfx1250) — matrix operand order repaired and checked by reading, not compiled, not run on hardware
//
#include <hip/hip_runtime.h>
#include <hip/hip_bf16.h>

typedef _Float16 v16h __attribute__((ext_vector_type(16)));
typedef float    v8f  __attribute__((ext_vector_type(8)));

#define NR     4096
#define GDIM   1024
#define DT_F   0.1f

// Matches reference _to_grid: trunc toward zero (pos >= 0), clamp to [0,1023].
// Single constant multiply instead of the precise-divide expansion.
__device__ __forceinline__ int to_grid(float p) {
    int g = (int)(p * (1024.0f / 100.0f));
    g = g < 0 ? 0 : g;
    return g > (GDIM - 1) ? (GDIM - 1) : g;
}

// ---- K0: normalize activities over full 64 dims, keep first 16 dims as f16 ----
__global__ void uprt_prep(const float* __restrict__ act, _Float16* __restrict__ h16) {
    int i = blockIdx.x * blockDim.x + threadIdx.x;
    if (i >= NR) return;
    const float* a = act + (size_t)i * 64;
    float s = 0.0f;
#pragma unroll
    for (int k = 0; k < 64; ++k) { float v = a[k]; s += v * v; }
    float inv = 1.0f / (sqrtf(s) + 1e-8f);
#pragma unroll
    for (int k = 0; k < 16; ++k) h16[i * 16 + k] = (_Float16)(a[k] * inv);
}

// ---- grid-stride copy with fused scale (scale=1 for plain copy) ----
__global__ void uprt_copyscale(const float4* __restrict__ src, float4* __restrict__ dst,
                               int n4, float scale) {
    int i = blockIdx.x * blockDim.x + threadIdx.x;
    int stride = gridDim.x * blockDim.x;
    for (; i < n4; i += stride) {
        float4 v = src[i];
        v.x *= scale; v.y *= scale; v.z *= scale; v.w *= scale;
        dst[i] = v;
    }
}

// ---- K2: consciousness scatter-add (one thread per robot-channel) ----
__global__ void uprt_scatter_c(const float* __restrict__ pos, const float* __restrict__ act,
                               float* __restrict__ c_tmp) {
    int idx = blockIdx.x * blockDim.x + threadIdx.x;
    if (idx >= NR * 16) return;
    int i = idx >> 4, ch = idx & 15;
    int gx = to_grid(pos[2 * i]);
    int gy = to_grid(pos[2 * i + 1]);
    atomicAdd(c_tmp + ((size_t)(gx * GDIM + gy)) * 16 + ch, act[(size_t)i * 64 + ch] * DT_F);
}

// ---- K3: genetic scatter-add with decay folded in ----
__global__ void uprt_scatter_g(const float* __restrict__ pos, const float* __restrict__ act,
                               float* __restrict__ g_out, float fg) {
    int idx = blockIdx.x * blockDim.x + threadIdx.x;
    if (idx >= NR * 32) return;
    int i = idx >> 5, ch = idx & 31;
    int gx = to_grid(pos[2 * i]);
    int gy = to_grid(pos[2 * i + 1]);
    atomicAdd(g_out + ((size_t)(gx * GDIM + gy)) * 32 + ch,
              act[(size_t)i * 64 + ch] * DT_F * 0.1f * fg);
}

// ---- K4: pairwise resonance, one wave per 16x16 tile of the upper triangle ----
// sim tile = A(16x16, K=16 zero-padded to 32) @ B via v_wmma_f32_16x16x32_f16
__global__ void __launch_bounds__(32) uprt_pairs(const float* __restrict__ pos,
                                                 const _Float16* __restrict__ h16,
                                                 float* __restrict__ r_tmp) {
    int ti = blockIdx.x;
    int tj = blockIdx.y;
    if (tj < ti) return;          // upper triangle tiles only (uniform per wave)

    int lane = threadIdx.x;       // wave32
    int m15  = lane & 15;
    int khalf = (lane >= 16) ? 8 : 0;

    // A fragment (16x32 f16): lanes 0-15 hold M=m15, K=0..15; lanes 16-31 hold
    // K=16..31 which stay zero (K padding).
    v16h A = {};
    {
        const _Float16* arow = h16 + ((size_t)(ti * 16 + m15)) * 16 + khalf;
#pragma unroll
        for (int k = 0; k < 16; ++k) if (lane < 16) A[k] = arow[k];
    }
    // B fragment (32x16 f16): lanes 0-15 hold N=lane, K=0..15; lanes 16-31 hold
    // K=16..31 which is all zero padding here.
    v16h B = {};
    if (lane < 16) {
        const _Float16* bcol = h16 + ((size_t)(tj * 16 + m15)) * 16;
#pragma unroll
        for (int k = 0; k < 16; ++k) B[k] = bcol[k];
    }

    v8f acc = {};
    acc = __builtin_amdgcn_wmma_f32_16x16x32_f16(false, A, false, B, (short)0, acc,
                                                 false, false);

    // C layout: element v of this lane is (M = v + (lane>=16 ? 8:0), N = lane&15)
    int j = tj * 16 + m15;
    float pjx = pos[2 * j], pjy = pos[2 * j + 1];
    int mbase = (lane >= 16) ? 8 : 0;
#pragma unroll
    for (int v = 0; v < 8; ++v) {
        int i = ti * 16 + v + mbase;
        if (j <= i) continue;                      // triu(k=1)
        float pix = pos[2 * i], piy = pos[2 * i + 1];
        float dx = pix - pjx, dy = piy - pjy;
        float d2 = dx * dx + dy * dy;
        if (d2 >= 100.0f) continue;                // dist < 10  <=>  d2 < 100
        float sim = acc[v];
        // sigmoid(5*sim) via v_exp + v_rcp (avoids the precise-divide expansion)
        float res = __builtin_amdgcn_rcpf(1.0f + __expf(-5.0f * sim));
        float contrib = res * DT_F;
        float mx = 0.5f * (pix + pjx), my = 0.5f * (piy + pjy);
        int mgx = to_grid(mx), mgy = to_grid(my);
        float* cell = r_tmp + ((size_t)(mgx * GDIM + mgy)) * 16;
#pragma unroll
        for (int ch = 0; ch < 16; ++ch) atomicAdd(cell + ch, contrib);
    }
}

// ---- K5: 5-point diffusion (replicate pad) + decay, float4 = 4 channels/thread ----
__global__ void uprt_diffuse(const float4* __restrict__ src, float4* __restrict__ dst,
                             float cdt, float decay) {
    int idx = blockIdx.x * blockDim.x + threadIdx.x;
    if (idx >= GDIM * GDIM * 4) return;
    int q = idx & 3;            // which group of 4 channels
    int cell = idx >> 2;
    int y = cell & (GDIM - 1);
    int x = cell >> 10;
    int xp = x + 1 < GDIM ? x + 1 : GDIM - 1;
    int xm = x > 0 ? x - 1 : 0;
    int yp = y + 1 < GDIM ? y + 1 : GDIM - 1;
    int ym = y > 0 ? y - 1 : 0;
    float4 c  = src[(size_t)(x * GDIM + y) * 4 + q];
    float4 n0 = src[(size_t)(xp * GDIM + y) * 4 + q];
    float4 n1 = src[(size_t)(xm * GDIM + y) * 4 + q];
    float4 n2 = src[(size_t)(x * GDIM + yp) * 4 + q];
    float4 n3 = src[(size_t)(x * GDIM + ym) * 4 + q];
    float4 o;
    o.x = (c.x + cdt * (n0.x + n1.x + n2.x + n3.x - 4.0f * c.x)) * decay;
    o.y = (c.y + cdt * (n0.y + n1.y + n2.y + n3.y - 4.0f * c.y)) * decay;
    o.z = (c.z + cdt * (n0.z + n1.z + n2.z + n3.z - 4.0f * c.z)) * decay;
    o.w = (c.w + cdt * (n0.w + n1.w + n2.w + n3.w - 4.0f * c.w)) * decay;
    dst[idx] = o;
}

extern "C" void kernel_launch(void* const* d_in, const int* in_sizes, int n_in,
                              void* d_out, int out_size, void* d_ws, size_t ws_size,
                              hipStream_t stream) {
    const float* pos = (const float*)d_in[0];   // [4096, 2]
    const float* act = (const float*)d_in[1];   // [4096, 64]
    const float* cf  = (const float*)d_in[2];   // [1024,1024,16]
    const float* rf  = (const float*)d_in[3];   // [1024,1024,16]
    const float* gf  = (const float*)d_in[4];   // [1024,1024,32]

    const size_t CE = (size_t)GDIM * GDIM * 16; // 16,777,216
    float* out_c = (float*)d_out;
    float* out_r = out_c + CE;
    float* out_g = out_r + CE;                  // 2*CE elements

    // Reuse the g output region (exactly 2*CE floats) as pre-diffusion scratch;
    // it is overwritten last.
    float* c_tmp = out_g;
    float* r_tmp = out_g + CE;

    _Float16* h16 = (_Float16*)d_ws;            // 4096*16 f16 = 128 KB

    // 0) normalized f16 activities for WMMA
    uprt_prep<<<16, 256, 0, stream>>>(act, h16);

    // 1) stage c and r fields into scratch (g region)
    int n4c = (int)(CE / 4);
    uprt_copyscale<<<8192, 256, 0, stream>>>((const float4*)cf, (float4*)c_tmp, n4c, 1.0f);
    uprt_copyscale<<<8192, 256, 0, stream>>>((const float4*)rf, (float4*)r_tmp, n4c, 1.0f);

    // 2) scatter robot activities into c_tmp
    uprt_scatter_c<<<(NR * 16) / 256, 256, 0, stream>>>(pos, act, c_tmp);

    // 3) pairwise resonance via WMMA, scatter midpoints into r_tmp
    uprt_pairs<<<dim3(256, 256), 32, 0, stream>>>(pos, h16, r_tmp);

    // 4) diffuse + decay into final c and r outputs
    int ndiff = GDIM * GDIM * 4;
    uprt_diffuse<<<(ndiff + 255) / 256, 256, 0, stream>>>((const float4*)c_tmp, (float4*)out_c,
                                                          0.1f * DT_F, 1.0f - 0.01f * DT_F);
    uprt_diffuse<<<(ndiff + 255) / 256, 256, 0, stream>>>((const float4*)r_tmp, (float4*)out_r,
                                                          0.2f * DT_F, 1.0f - 0.005f * DT_F);

    // 5) genetic field: decay-scaled copy (overwrites scratch), then scatter
    const float fg = 1.0f - 0.01f * 0.1f * DT_F;   // 0.9999
    int n4g = (int)(2 * CE / 4);
    uprt_copyscale<<<8192, 256, 0, stream>>>((const float4*)gf, (float4*)out_g, n4g, fg);
    uprt_scatter_g<<<(NR * 32) / 256, 256, 0, stream>>>(pos, act, out_g, fg);
}
